// MultiHeadSelfAttention_89266600280710
// MI455X (gfx1250) — matrix operand order repaired and checked by reading, not compiled, not run on hardware
//
#include <hip/hip_runtime.h>

typedef __attribute__((ext_vector_type(16))) _Float16 v16h;
typedef __attribute__((ext_vector_type(8)))  _Float16 v8h;
typedef __attribute__((ext_vector_type(8)))  float    v8f;

#define DEV static __device__ __forceinline__

// Build a 16-half A-operand from two 16B chunks (A-layout: lane holds
// row = lane%16; halves = K[k0+hs*16 .. +7] then K[k0+hs*16+8 .. +7]).
DEV v16h make_a(const _Float16* p0, const _Float16* p1) {
  v8h lo = *(const v8h*)p0;
  v8h hi = *(const v8h*)p1;
  return __builtin_shufflevector(lo, hi, 0,1,2,3,4,5,6,7,8,9,10,11,12,13,14,15);
}

DEV v8f wmma_f16(v16h a, v16h b, v8f c) {
  return __builtin_amdgcn_wmma_f32_16x16x32_f16(false, a, false, b, (short)0, c,
                                                false, false);
}

// Async DMA: 16B per lane, global -> LDS, GV mode (no SADDR). ASYNCcnt-tracked.
DEV void async_b128(unsigned lds_byte_off, const void* gaddr) {
  asm volatile("global_load_async_to_lds_b128 %0, %1, off"
               :: "v"(lds_byte_off), "v"((unsigned long long)(size_t)gaddr)
               : "memory");
}

// ---------------------------------------------------------------- f32 -> f16
__global__ void cvt_f32_f16(const float* __restrict__ src,
                            _Float16* __restrict__ dst, int n) {
  int i = blockIdx.x * blockDim.x + threadIdx.x;
  if (i < n) dst[i] = (_Float16)src[i];
}

// ------------------------------------------------------ GEMM: C = A * W^T
// A: [M,K] f16 row-major.  W: [N,K] f16 row-major (W^T col n == W row n).
// Block (8 waves) computes a 128x64 tile; per 32-k step it double-buffers
// A(128x32) + B(64x32) through LDS via global_load_async_to_lds_b128.
// mode 0/1: store f16 [B,H,S,64] (Q scaled / K)
// mode 2  : store f16 [B,H,64,S] (V transposed)
// mode 3  : store f32 [M,N]      (final output)
__global__ void __launch_bounds__(256)
gemm16(const _Float16* __restrict__ A, const _Float16* __restrict__ W,
       int M, int N, int K, int mode, float scaleOut, void* __restrict__ outp,
       int Hn, int Sn) {
  __shared__ __align__(64) _Float16 Als[2][128 * 32];   // 8KB x2
  __shared__ __align__(64) _Float16 Bls[2][64 * 32];    // 4KB x2

  int tid   = threadIdx.x;
  int lane  = tid & 31;
  int wslot = tid >> 5;                       // 0..7: M sub-tile of this wave
  int Nb = N >> 6;
  int mb = (int)blockIdx.x / Nb;              // 128-row block index
  int nb = (int)blockIdx.x - mb * Nb;         // 64-col block index
  if (mb * 128 >= M) return;                  // uniform per block
  int hs = lane >> 4, r = lane & 15;

  const _Float16* Ag = A + (size_t)mb * 128 * K;   // block A base
  const _Float16* Wg = W + (size_t)nb * 64 * K;    // block W base

  int sub   = lane >> 2;          // 0..7: row within a 512B DMA chunk
  int inrow = (lane & 3) * 16;    // byte offset within a 64B row

  // Issue one k-step's tile DMA into buffer `buf` (3 async b128 per wave).
  auto issue = [&](int buf, int k0) {
#pragma unroll
    for (int i = 0; i < 2; ++i) {             // A: chunks 2*wslot, 2*wslot+1
      int c = wslot * 2 + i;
      int row = c * 8 + sub;
      unsigned loff = (unsigned)(size_t)&Als[buf][0] + (unsigned)(c * 512 + lane * 16);
      const char* g = (const char*)(Ag + (size_t)row * K + k0) + inrow;
      async_b128(loff, g);
    }
    {                                         // B: chunk `wslot`
      int row = wslot * 8 + sub;
      unsigned loff = (unsigned)(size_t)&Bls[buf][0] + (unsigned)(wslot * 512 + lane * 16);
      const char* g = (const char*)(Wg + (size_t)row * K + k0) + inrow;
      async_b128(loff, g);
    }
  };

  v8f acc[4] = {{}, {}, {}, {}};
  int ksteps = K >> 5;

  issue(0, 0);
  for (int ks = 0; ks < ksteps; ++ks) {
    int cur = ks & 1;
    bool more = (ks + 1) < ksteps;
    if (more) issue(cur ^ 1, (ks + 1) * 32);
    // Drain this wave's DMAs for `cur` (3 newer ones may stay in flight),
    // then publish block-wide.
    if (more) asm volatile("s_wait_asynccnt 3" ::: "memory");
    else      asm volatile("s_wait_asynccnt 0" ::: "memory");
    __syncthreads();

    const _Float16* arow = &Als[cur][0] + (wslot * 16 + r) * 32;
    v16h a = make_a(arow + hs * 16, arow + hs * 16 + 8);
#pragma unroll
    for (int t = 0; t < 4; ++t) {
      const _Float16* brow = &Bls[cur][0] + (t * 16 + r) * 32 + hs * 16;
      acc[t] = wmma_f16(a, *(const v16h*)brow, acc[t]);
    }
    __syncthreads();   // all reads of `cur` done before it is refilled
  }

#pragma unroll
  for (int t = 0; t < 4; ++t) {
    int n = nb * 64 + t * 16 + r;
#pragma unroll
    for (int g = 0; g < 8; ++g) {
      int m = mb * 128 + wslot * 16 + g + 8 * hs;
      float v = acc[t][g] * scaleOut;
      if (mode == 3) {
        ((float*)outp)[(size_t)m * N + n] = v;
      } else {
        int b = m / Sn, s = m - b * Sn;
        int h = n >> 6, dk = n & 63;
        size_t idx;
        if (mode == 2) idx = (((size_t)(b * Hn + h)) * 64 + dk) * Sn + s;
        else           idx = (((size_t)(b * Hn + h)) * Sn + s) * 64 + dk;
        ((_Float16*)outp)[idx] = (_Float16)v;
      }
    }
  }
}

// ----------------------------------------------- flash attention (causal)
// One wave per 16-query tile. Q pre-scaled by 1/sqrt(dk). dk = 64.
// Q,K: [B*H, S, 64] f16.  Vt: [B*H, 64, S] f16.  O: [B, S, H*64] f16.
__global__ void __launch_bounds__(128)
flash16(const _Float16* __restrict__ Q, const _Float16* __restrict__ Km,
        const _Float16* __restrict__ Vt, _Float16* __restrict__ O,
        int B, int H, int S) {
  __shared__ __align__(16) _Float16 pbuf[4][16][32];
  int lane  = threadIdx.x & 31;
  int wslot = threadIdx.x >> 5;
  int qtiles = S >> 4;
  int gwave = blockIdx.x * 4 + wslot;
  if (gwave >= B * H * qtiles) return;
  int bh = gwave / qtiles;
  int qi = gwave - bh * qtiles;
  int q0 = qi << 4;
  int hs = lane >> 4, r = lane & 15;

  const _Float16* Qb = Q  + (size_t)bh * S * 64;
  const _Float16* Kb = Km + (size_t)bh * S * 64;
  const _Float16* Vb = Vt + (size_t)bh * 64 * S;

  const _Float16* qrow = Qb + (size_t)(q0 + r) * 64;
  v16h qa0 = make_a(qrow + hs * 16,      qrow + 8 + hs * 16);
  v16h qa1 = make_a(qrow + 32 + hs * 16, qrow + 40 + hs * 16);

  v8f o0 = {}, o1 = {}, o2 = {}, o3 = {};
  float rmax[8], rsum[8];
#pragma unroll
  for (int g = 0; g < 8; ++g) { rmax[g] = -1e30f; rsum[g] = 0.f; }

  int kend = q0 + 16;                 // causal: keys 0 .. q0+15
  for (int kb = 0; kb < kend; kb += 32) {
    bool t1 = (kb + 16) < kend;
    v8f s0 = {}, s1 = {};
    {
      const _Float16* kp = Kb + (size_t)(kb + r) * 64 + hs * 16;
      s0 = wmma_f16(qa0, *(const v16h*)kp, s0);
      s0 = wmma_f16(qa1, *(const v16h*)(kp + 32), s0);
    }
    if (t1) {
      const _Float16* kp = Kb + (size_t)(kb + 16 + r) * 64 + hs * 16;
      s1 = wmma_f16(qa0, *(const v16h*)kp, s1);
      s1 = wmma_f16(qa1, *(const v16h*)(kp + 32), s1);
    }
    // stream the next K tile (global_prefetch_b8; speculative, OOB-safe)
    __builtin_prefetch(Kb + (size_t)(kb + 32 + r) * 64, 0, 1);

    // causal mask (C-layout: row = q0 + g + 8*hs, key = kb + t*16 + r)
    float tval[8];
#pragma unroll
    for (int g = 0; g < 8; ++g) {
      int m = q0 + g + 8 * hs;
      if (kb + r > m)                s0[g] = -1e30f;
      if (!t1 || (kb + 16 + r > m))  s1[g] = -1e30f;
      tval[g] = fmaxf(s0[g], s1[g]);
    }
    // row-max reduce across the 16 lanes holding each row (half-wave)
#pragma unroll
    for (int off = 8; off; off >>= 1)
#pragma unroll
      for (int g = 0; g < 8; ++g)
        tval[g] = fmaxf(tval[g], __shfl_xor(tval[g], off, 32));

    float alpha[8];
#pragma unroll
    for (int g = 0; g < 8; ++g) {
      float mn = fmaxf(rmax[g], tval[g]);
      alpha[g] = __expf(rmax[g] - mn);
      rmax[g] = mn;
    }
#pragma unroll
    for (int g = 0; g < 8; ++g) {
      s0[g] = __expf(s0[g] - rmax[g]);
      s1[g] = __expf(s1[g] - rmax[g]);
      tval[g] = s0[g] + s1[g];
    }
#pragma unroll
    for (int off = 8; off; off >>= 1)
#pragma unroll
      for (int g = 0; g < 8; ++g)
        tval[g] += __shfl_xor(tval[g], off, 32);
#pragma unroll
    for (int g = 0; g < 8; ++g) {
      rsum[g] = rsum[g] * alpha[g] + tval[g];
      o0[g] *= alpha[g]; o1[g] *= alpha[g];
      o2[g] *= alpha[g]; o3[g] *= alpha[g];
    }

    // Transpose P from C-layout to A-layout via per-wave LDS tile.
#pragma unroll
    for (int g = 0; g < 8; ++g) {
      int row = g + 8 * hs;
      pbuf[wslot][row][r]      = (_Float16)s0[g];
      pbuf[wslot][row][16 + r] = (_Float16)s1[g];
    }
    asm volatile("s_wait_dscnt 0" ::: "memory");
    const _Float16* prow = &pbuf[wslot][r][0];
    v16h pa = make_a(prow + hs * 16, prow + hs * 16 + 8);

    // PV: B-operand = 16 consecutive keys of one dk column, contiguous in Vt.
    int kv = kb + hs * 16;
    if (kv > S - 16) kv = S - 16;     // masked half (P==0): keep load in-bounds
    const _Float16* vp = Vb + (size_t)r * S + kv;
    o0 = wmma_f16(pa, *(const v16h*)vp, o0);
    o1 = wmma_f16(pa, *(const v16h*)(vp + (size_t)16 * S), o1);
    o2 = wmma_f16(pa, *(const v16h*)(vp + (size_t)32 * S), o2);
    o3 = wmma_f16(pa, *(const v16h*)(vp + (size_t)48 * S), o3);
  }

  // normalize and store into O [B, S, H*64]
  int b = bh / H, h = bh - b * H;
  _Float16* Ob = O + (size_t)b * S * (size_t)(H * 64) + (size_t)h * 64;
#pragma unroll
  for (int g = 0; g < 8; ++g) {
    float inv = 1.0f / rsum[g];
    int s = q0 + g + 8 * hs;
    _Float16* orow = Ob + (size_t)s * (H * 64);
    orow[r]      = (_Float16)(o0[g] * inv);
    orow[16 + r] = (_Float16)(o1[g] * inv);
    orow[32 + r] = (_Float16)(o2[g] * inv);
    orow[48 + r] = (_Float16)(o3[g] * inv);
  }
}

extern "C" void kernel_launch(void* const* d_in, const int* in_sizes, int n_in,
                              void* d_out, int out_size, void* d_ws, size_t ws_size,
                              hipStream_t stream) {
  if (n_in < 7) return;
  const float* Wq = (const float*)d_in[2];
  const float* Wk = (const float*)d_in[3];
  const float* Wv = (const float*)d_in[4];
  const float* Wo = (const float*)d_in[5];
  const float* X  = (const float*)d_in[6];

  const int D = 1024, H = 16, S = 2048;
  int M = in_sizes[6] / D;   // B*S tokens
  int B = M / S;

  // workspace layout (f16): Xh | Wq | Wk | Wv | Wo | Q | K | Vt | O
  size_t nX = (size_t)M * D;
  size_t nW = (size_t)D * D;
  _Float16* Xh  = (_Float16*)d_ws;
  _Float16* Wqh = Xh  + nX;
  _Float16* Wkh = Wqh + nW;
  _Float16* Wvh = Wkh + nW;
  _Float16* Woh = Wvh + nW;
  _Float16* Qh  = Woh + nW;
  _Float16* Kh  = Qh  + nX;
  _Float16* Vth = Kh  + nX;
  _Float16* Oh  = Vth + nX;

  const int thr = 256;
  cvt_f32_f16<<<(int)((nX + thr - 1) / thr), thr, 0, stream>>>(X,  Xh,  (int)nX);
  cvt_f32_f16<<<(int)((nW + thr - 1) / thr), thr, 0, stream>>>(Wq, Wqh, (int)nW);
  cvt_f32_f16<<<(int)((nW + thr - 1) / thr), thr, 0, stream>>>(Wk, Wkh, (int)nW);
  cvt_f32_f16<<<(int)((nW + thr - 1) / thr), thr, 0, stream>>>(Wv, Wvh, (int)nW);
  cvt_f32_f16<<<(int)((nW + thr - 1) / thr), thr, 0, stream>>>(Wo, Woh, (int)nW);

  int gemmBlocks = (M / 128) * (D / 64);    // 128x64 tile per block
  gemm16<<<gemmBlocks, 256, 0, stream>>>(Xh, Wqh, M, D, D, 0, 0.125f, Qh,  H, S);
  gemm16<<<gemmBlocks, 256, 0, stream>>>(Xh, Wkh, M, D, D, 1, 1.0f,   Kh,  H, S);
  gemm16<<<gemmBlocks, 256, 0, stream>>>(Xh, Wvh, M, D, D, 2, 1.0f,   Vth, H, S);

  int fwaves = B * H * (S / 16);
  flash16<<<(fwaves + 3) / 4, 128, 0, stream>>>(Qh, Kh, Vth, Oh, B, H, S);

  gemm16<<<gemmBlocks, 256, 0, stream>>>(Oh, Woh, M, D, D, 3, 1.0f, d_out, H, S);
}
